// task_loss_84705345012103
// MI455X (gfx1250) — matrix operand order repaired and checked by reading, not compiled, not run on hardware
//
#include <hip/hip_runtime.h>

// ---------------------------------------------------------------------------
// Problem: B=32768, N=512.
//   r = exp(sigma - log_sigma_hat);  Gamma SPD well-conditioned (kappa~5)
//   v = Gamma^{-1} SR  (u == v by symmetry)  -> solved with 20-iter CG
//   t = r*v;  first = t.SR/k;  second = t^T Gamma t/(2k);  out = mean(second-first)
// Hot spot: Y = T @ Gamma^T  (B x N x N GEMM, 17.2 GFLOP) -> bf16 WMMA with
// split-precision (hi+lo) to get ~f32 accuracy at tensor-core rate.
// ---------------------------------------------------------------------------

typedef __attribute__((ext_vector_type(16))) __bf16        v16bf;
typedef __attribute__((ext_vector_type(8)))  float         v8f;
typedef __attribute__((ext_vector_type(4)))  float         v4f;
typedef __attribute__((ext_vector_type(4)))  unsigned int  v4u;

#define NN     512
#define MTILE  32
#define BLOCK  256

union BfFrag { v4u q[2]; v16bf v; };

__device__ __forceinline__ unsigned short f2bf(float f) {
    unsigned u = __float_as_uint(f);
    unsigned r = (u + 0x7FFFu + ((u >> 16) & 1u)) >> 16;   // round-to-nearest-even
    return (unsigned short)r;
}
__device__ __forceinline__ float bf2f(unsigned short s) {
    return __uint_as_float(((unsigned)s) << 16);
}

// ---------------------------------------------------------------------------
// Kernel 1: CG solve Gamma x = SR.  One WG, 1024 threads (2 threads per row).
// Fixed 20 iterations (kappa ~ 5 -> residual ~4e-9), deterministic reductions.
// ---------------------------------------------------------------------------
__global__ __launch_bounds__(1024) void cg_kernel(
    const float* __restrict__ G, const float* __restrict__ SR,
    float* __restrict__ vout)
{
    __shared__ __align__(16) float sP[NN], sRv[NN], sX[NN], sAp[NN];
    __shared__ float sRed[1024];

    const int t = threadIdx.x;
    const int i = t >> 1;          // row
    const int h = t & 1;           // half of the row

    if (t < NN) { float s = SR[t]; sX[t] = 0.f; sRv[t] = s; sP[t] = s; }
    __syncthreads();

    float rs;
    {
        float loc = (t < NN) ? sRv[t] * sRv[t] : 0.f;
        sRed[t] = loc; __syncthreads();
        for (int s = 512; s > 0; s >>= 1) { if (t < s) sRed[t] += sRed[t + s]; __syncthreads(); }
        rs = sRed[0]; __syncthreads();
    }

    for (int it = 0; it < 20; ++it) {
        // Ap = Gamma * p   (each thread: 256 MACs over its half-row, f32x4)
        const v4f* grow = reinterpret_cast<const v4f*>(G + (long)i * NN + h * 256);
        const v4f* pvec = reinterpret_cast<const v4f*>(&sP[h * 256]);
        float s = 0.f;
        #pragma unroll 8
        for (int j = 0; j < 64; ++j) {
            v4f g = grow[j], p = pvec[j];
            s += g.x * p.x + g.y * p.y + g.z * p.z + g.w * p.w;
        }
        sRed[t] = s; __syncthreads();

        float pAp_loc = 0.f;
        if (t < NN) {
            float ap = sRed[2 * t] + sRed[2 * t + 1];
            sAp[t] = ap;
            pAp_loc = sP[t] * ap;
        }
        __syncthreads();
        sRed[t] = (t < NN) ? pAp_loc : 0.f; __syncthreads();
        for (int s2 = 512; s2 > 0; s2 >>= 1) { if (t < s2) sRed[t] += sRed[t + s2]; __syncthreads(); }
        const float alpha = rs / sRed[0];
        __syncthreads();

        float rr_loc = 0.f;
        if (t < NN) {
            sX[t] += alpha * sP[t];
            float nr = sRv[t] - alpha * sAp[t];
            sRv[t] = nr;
            rr_loc = nr * nr;
        }
        __syncthreads();
        sRed[t] = (t < NN) ? rr_loc : 0.f; __syncthreads();
        for (int s2 = 512; s2 > 0; s2 >>= 1) { if (t < s2) sRed[t] += sRed[t + s2]; __syncthreads(); }
        const float rsn = sRed[0];
        __syncthreads();

        const float beta = rsn / rs;
        if (t < NN) sP[t] = sRv[t] + beta * sP[t];
        rs = rsn;
        __syncthreads();
    }
    if (t < NN) vout[t] = sX[t];
}

// ---------------------------------------------------------------------------
// Kernel 2: split Gamma into bf16 hi/lo, stored in WMMA B-fragment order:
//   GB[((j>>5)*512 + i)*32 + ((j>>3)&1)*16 + ((j>>4)&1)*8 + (j&7)] = Gamma[i][j]  (= Gamma^T)
// so each lane's 16-element B fragment is 32 contiguous bytes (2x b128 loads).
// ---------------------------------------------------------------------------
__global__ __launch_bounds__(BLOCK) void split_kernel(
    const float* __restrict__ G,
    unsigned short* __restrict__ ghi, unsigned short* __restrict__ glo)
{
    const int e = blockIdx.x * BLOCK + threadIdx.x;
    if (e < NN * NN) {
        const int i = e >> 9, j = e & (NN - 1);
        const float g = G[e];
        const unsigned short hi = f2bf(g);
        const unsigned short lo = f2bf(g - bf2f(hi));
        const int dst = ((j >> 5) * NN + i) * 32 + ((j >> 3) & 1) * 16 + ((j >> 4) & 1) * 8 + (j & 7);
        ghi[dst] = hi;
        glo[dst] = lo;
    }
}

// ---------------------------------------------------------------------------
// Kernel 3: fused main pass.  32 batch rows per WG, 8 waves.
//  Phase A: t = exp(sg-lsh)*v, split to bf16 hi/lo in LDS (A-fragment layout),
//           accumulate first = t.SR.
//  Phase B: Y = T Gamma^T via 3 split bf16 WMMAs per tile (hi*hi+hi*lo+lo*hi),
//           fold second = sum(Y*t) with shuffle reductions.
// ---------------------------------------------------------------------------
__global__ __launch_bounds__(BLOCK) void fused_kernel(
    const float* __restrict__ lsh, const float* __restrict__ sg,
    const float* __restrict__ SR,  const float* __restrict__ vvec,
    const unsigned short* __restrict__ gbhi,
    const unsigned short* __restrict__ gblo,
    const int* __restrict__ kptr, float* __restrict__ partials)
{
    __shared__ unsigned short sThi[MTILE * NN];   // 32 KB
    __shared__ unsigned short sTlo[MTILE * NN];   // 32 KB
    __shared__ float sSR[NN], sV[NN];
    __shared__ float sFp[MTILE * 8];
    __shared__ float sSec[8 * MTILE];

    const int tid = threadIdx.x;
    const long b0 = (long)blockIdx.x * MTILE;

    for (int i = tid; i < NN; i += BLOCK) { sSR[i] = SR[i]; sV[i] = vvec[i]; }
    __syncthreads();

    // ---------------- Phase A: elementwise + split + first-term dot ---------
    {
        const int row = tid >> 3;              // 8 threads per row
        const int c0  = (tid & 7) * 64;        // 64 columns per thread
        const v4f* lp = reinterpret_cast<const v4f*>(lsh + (b0 + row) * NN + c0);
        const v4f* spv = reinterpret_cast<const v4f*>(sg + (b0 + row) * NN + c0);
        float fp = 0.f;
        #pragma unroll 4
        for (int it = 0; it < 16; ++it) {
            v4f a = lp[it], b = spv[it];
            #pragma unroll
            for (int c = 0; c < 4; ++c) {
                const int j = c0 + it * 4 + c;
                const float r = __expf(b[c] - a[c]);     // v_exp_f32
                const float t = r * sV[j];
                const unsigned short hi = f2bf(t);
                const unsigned short lo = f2bf(t - bf2f(hi));
                sThi[row * NN + j] = hi;
                sTlo[row * NN + j] = lo;
                fp += t * sSR[j];
            }
        }
        sFp[row * 8 + (tid & 7)] = fp;
    }
    __syncthreads();

    // ---------------- Phase B: split-bf16 WMMA GEMM + quadratic form --------
    const int wave  = tid >> 5;
    const int lane  = tid & 31;
    const int ln    = lane & 15;
    const int half  = lane >> 4;
    const int nbase = wave * 64;               // 4 n-tiles of 16 per wave

    v8f acc[2][4];
    {
        const v8f z = {0.f, 0.f, 0.f, 0.f, 0.f, 0.f, 0.f, 0.f};
        #pragma unroll
        for (int m = 0; m < 2; ++m)
            #pragma unroll
            for (int nt = 0; nt < 4; ++nt) acc[m][nt] = z;
    }

    for (int ks = 0; ks < 16; ++ks) {          // K = 512 in steps of 32
        BfFrag bh[4], bl[4];
        #pragma unroll
        for (int nt = 0; nt < 4; ++nt) {
            const int n = nbase + nt * 16 + ln;
            const v4u* ph = reinterpret_cast<const v4u*>(gbhi + ((ks * NN + n) * 32 + half * 16));
            const v4u* pl = reinterpret_cast<const v4u*>(gblo + ((ks * NN + n) * 32 + half * 16));
            bh[nt].q[0] = ph[0]; bh[nt].q[1] = ph[1];
            bl[nt].q[0] = pl[0]; bl[nt].q[1] = pl[1];
        }
        #pragma unroll
        for (int m = 0; m < 2; ++m) {
            // A fragment: row = m*16 + ln ; K elems {c..c+7, c+16..c+23}, c=8*half
            const int abase = (m * 16 + ln) * NN + ks * 32 + half * 8;
            BfFrag ah, al;
            ah.q[0] = *reinterpret_cast<const v4u*>(&sThi[abase]);
            ah.q[1] = *reinterpret_cast<const v4u*>(&sThi[abase + 16]);
            al.q[0] = *reinterpret_cast<const v4u*>(&sTlo[abase]);
            al.q[1] = *reinterpret_cast<const v4u*>(&sTlo[abase + 16]);
            #pragma unroll
            for (int nt = 0; nt < 4; ++nt) {
                acc[m][nt] = __builtin_amdgcn_wmma_f32_16x16x32_bf16(
                    false, ah.v, false, bh[nt].v, (short)0, acc[m][nt], false, false);
                acc[m][nt] = __builtin_amdgcn_wmma_f32_16x16x32_bf16(
                    false, ah.v, false, bl[nt].v, (short)0, acc[m][nt], false, false);
                acc[m][nt] = __builtin_amdgcn_wmma_f32_16x16x32_bf16(
                    false, al.v, false, bh[nt].v, (short)0, acc[m][nt], false, false);
            }
        }
    }

    // second[b] partials: Y(m,n) * t(m,n), D layout: M = rr + 8*half, N = ln
    float secacc[2][8];
    #pragma unroll
    for (int m = 0; m < 2; ++m)
        #pragma unroll
        for (int rr = 0; rr < 8; ++rr) secacc[m][rr] = 0.f;

    #pragma unroll
    for (int m = 0; m < 2; ++m)
        #pragma unroll
        for (int nt = 0; nt < 4; ++nt) {
            const int n = nbase + nt * 16 + ln;
            #pragma unroll
            for (int rr = 0; rr < 8; ++rr) {
                const int rowl = m * 16 + rr + 8 * half;
                const float t = bf2f(sThi[rowl * NN + n]) + bf2f(sTlo[rowl * NN + n]);
                secacc[m][rr] += acc[m][nt][rr] * t;
            }
        }

    // reduce across the 16 lanes that share the same rows
    #pragma unroll
    for (int m = 0; m < 2; ++m)
        #pragma unroll
        for (int rr = 0; rr < 8; ++rr) {
            float v = secacc[m][rr];
            v += __shfl_xor(v, 1);
            v += __shfl_xor(v, 2);
            v += __shfl_xor(v, 4);
            v += __shfl_xor(v, 8);
            secacc[m][rr] = v;
        }
    if (ln == 0) {
        #pragma unroll
        for (int m = 0; m < 2; ++m)
            #pragma unroll
            for (int rr = 0; rr < 8; ++rr)
                sSec[wave * MTILE + m * 16 + rr + 8 * half] = secacc[m][rr];
    }
    __syncthreads();

    // per-row combine + per-WG reduction (wave 0 only) -> deterministic
    if (tid < MTILE) {
        float sec = 0.f;
        #pragma unroll
        for (int w = 0; w < 8; ++w) sec += sSec[w * MTILE + tid];
        float fst = 0.f;
        #pragma unroll
        for (int s = 0; s < 8; ++s) fst += sFp[tid * 8 + s];
        const float kk = (float)(*kptr);
        float val = sec * (0.5f / kk) - fst * (1.0f / kk);   // second - first
        #pragma unroll
        for (int off = 1; off < 32; off <<= 1) val += __shfl_xor(val, off);
        if (tid == 0) partials[blockIdx.x] = val;
    }
}

// ---------------------------------------------------------------------------
// Kernel 4: final reduction -> -utils.mean() = sum(second-first)/B
// ---------------------------------------------------------------------------
__global__ __launch_bounds__(BLOCK) void reduce_kernel(
    const float* __restrict__ partials, int n, float invB, float* __restrict__ out)
{
    __shared__ float sred[BLOCK];
    const int t = threadIdx.x;
    float s = 0.f;
    for (int i = t; i < n; i += BLOCK) s += partials[i];
    sred[t] = s; __syncthreads();
    for (int k = BLOCK / 2; k > 0; k >>= 1) { if (t < k) sred[t] += sred[t + k]; __syncthreads(); }
    if (t == 0) out[0] = sred[0] * invB;
}

// ---------------------------------------------------------------------------
extern "C" void kernel_launch(void* const* d_in, const int* in_sizes, int n_in,
                              void* d_out, int out_size, void* d_ws, size_t ws_size,
                              hipStream_t stream)
{
    const float* lsh = (const float*)d_in[0];   // log_sigma_hat [B,N]
    const float* sg  = (const float*)d_in[1];   // sigma         [B,N]
    const float* SR  = (const float*)d_in[2];   // SR            [N]
    const float* Gam = (const float*)d_in[3];   // Gamma         [N,N]
    const int*   kp  = (const int*)d_in[4];     // k (scalar)
    float* out = (float*)d_out;

    const int Btot = in_sizes[0] / NN;          // 32768
    const int nblk = Btot / MTILE;              // 1024

    // ws layout: v (2 KB) | partials (4 KB) | pad | GB_hi (512 KB) | GB_lo (512 KB)
    char* ws = (char*)d_ws;
    float*          vvec     = (float*)ws;
    float*          partials = (float*)(ws + 2048);
    unsigned short* gbhi     = (unsigned short*)(ws + 8192);
    unsigned short* gblo     = (unsigned short*)(ws + 8192 + (size_t)NN * NN * sizeof(unsigned short));

    hipLaunchKernelGGL(cg_kernel,    dim3(1),             dim3(1024), 0, stream, Gam, SR, vvec);
    hipLaunchKernelGGL(split_kernel, dim3((NN * NN) / BLOCK), dim3(BLOCK), 0, stream, Gam, gbhi, gblo);
    hipLaunchKernelGGL(fused_kernel, dim3(nblk),          dim3(BLOCK), 0, stream,
                       lsh, sg, SR, vvec, gbhi, gblo, kp, partials);
    hipLaunchKernelGGL(reduce_kernel, dim3(1),            dim3(BLOCK), 0, stream,
                       partials, nblk, 1.0f / (float)Btot, out);
}
